// TransformerDecoderLayer_76287209111769
// MI455X (gfx1250) — matrix operand order repaired and checked by reading, not compiled, not run on hardware
//
#include <hip/hip_runtime.h>

// ---------------- problem dims ----------------
#define D_MODEL 1024
#define NHEAD   16
#define HEADD   64
#define DFF     4096
#define SEQL    1024
#define SEQS    1024
#define BATCH   4
#define NTOK    (SEQL * BATCH)   // 4096 rows (tgt side)
#define NMEM    (SEQS * BATCH)   // 4096 rows (memory side)

typedef unsigned short u16;
typedef unsigned int   u32;
typedef __attribute__((ext_vector_type(16))) __bf16 v16bf;
typedef __attribute__((ext_vector_type(8)))  float  v8f;

struct B32B { uint4 lo, hi; };

// float -> bf16 bits, round-to-nearest-even
__device__ inline u16 f2bf(float f) {
  u32 u = __builtin_bit_cast(u32, f);
  u32 r = u + 0x7FFFu + ((u >> 16) & 1u);
  return (u16)(r >> 16);
}

// A-fragment (16x32 bf16): lane r(=lane&15),c(=lane>>4) holds K = c*8..+7 and 16+c*8..+7
__device__ inline v16bf load_frag_a(const u16* p) {
  B32B t;
  t.lo = *(const uint4*)(p);
  t.hi = *(const uint4*)(p + 16);
  return __builtin_bit_cast(v16bf, t);
}
// B-fragment (32x16 bf16): lane n(=lane&15),c holds K = c*8..+7 and 16+c*8..+7 (row n of W)
__device__ inline v16bf load_frag_b(const u16* p) {
  B32B t;
  t.lo = *(const uint4*)(p);
  t.hi = *(const uint4*)(p + 16);
  return __builtin_bit_cast(v16bf, t);
}
__device__ inline v8f zero8() { v8f z; for (int e = 0; e < 8; ++e) z[e] = 0.f; return z; }

__device__ inline v8f wmma_bf16(v16bf a, v16bf b, v8f c) {
  return __builtin_amdgcn_wmma_f32_16x16x32_bf16(false, a, false, b, (short)0, c, false, false);
}

// ---------------- Tensor Data Mover (gfx1250) ----------------
#if __has_builtin(__builtin_amdgcn_tensor_load_to_lds)
#define HAVE_TDM 1
typedef __attribute__((ext_vector_type(4))) unsigned int v4u;
typedef __attribute__((ext_vector_type(8))) int v8i_;
typedef __attribute__((ext_vector_type(4))) int v4i_;

// 2D tile load: `rows` x `row_elems` bf16 tile at gptr (row stride ld elems) -> LDS (packed rows)
__device__ inline void tdm_load_2d(const u16* gptr, u16* lptr,
                                   int rows, int row_elems, int ld) {
  unsigned long long ga = (unsigned long long)(uintptr_t)gptr;
  unsigned lds = (unsigned)(uintptr_t)lptr;   // flat LDS addr: low 32 bits = WG-relative offset
  v4u g0;
  g0[0] = 1u;                                           // count = 1 valid descriptor
  g0[1] = lds;                                          // lds_addr
  g0[2] = (unsigned)(ga & 0xFFFFFFFFu);                 // global_addr lo
  g0[3] = (unsigned)((ga >> 32) & 0x1FFFFFFu) | (2u << 30);  // global_addr hi | type=2
  v8i_ g1;
  g1[0] = (int)(1u << 16);                              // data_size = 1 (2 bytes/elem)
  g1[1] = (int)(((unsigned)ld & 0xFFFFu) << 16);        // tensor_dim0[15:0]
  g1[2] = (int)((((unsigned)ld >> 16) & 0xFFFFu) |
                (((unsigned)rows & 0xFFFFu) << 16));    // tensor_dim0[31:16] | tensor_dim1[15:0]
  g1[3] = (int)(((unsigned)row_elems & 0xFFFFu) << 16); // tensor_dim1[31:16]=0 | tile_dim0
  g1[4] = (int)((unsigned)rows & 0xFFFFu);              // tile_dim1 | tile_dim2=0
  g1[5] = (int)(unsigned)ld;                            // tensor_dim0_stride[31:0]
  g1[6] = 0;                                            // stride hi | dim1_stride lo
  g1[7] = 0;
  v4i_ z4; z4[0] = z4[1] = z4[2] = z4[3] = 0;
  v8i_ z8; for (int i = 0; i < 8; ++i) z8[i] = 0;
  __builtin_amdgcn_tensor_load_to_lds(g0, g1, z4, z4, z8, 0);
}
#else
#define HAVE_TDM 0
#endif

// ---------------- LSQ 4-bit weight fake-quant -> bf16 ----------------
__global__ __launch_bounds__(256) void quant_kernel(const float* __restrict__ w,
                                                    const float* __restrict__ alphas, int aidx,
                                                    u16* __restrict__ out, int n) {
  int i = blockIdx.x * 256 + threadIdx.x;
  if (i >= n) return;
  float a = alphas[aidx];
  float q = w[i] / a;
  q = fminf(fmaxf(q, -8.f), 7.f);
  q = rintf(q) * a;
  out[i] = f2bf(q);
}

// ---------------- elementwise add + bf16 cast ----------------
__global__ __launch_bounds__(256) void addcast_kernel(const float* __restrict__ x,
                                                      const float* __restrict__ y,
                                                      u16* __restrict__ o, int n) {
  int i = blockIdx.x * 256 + threadIdx.x;
  if (i >= n) return;
  float t = x[i] + (y ? y[i] : 0.f);
  o[i] = f2bf(t);
}

// ---------------- LayerNorm (one row per block), bf16 out, optional (+add) second out ----------------
__global__ __launch_bounds__(256) void ln_kernel(const float* __restrict__ x,
                                                 const float* __restrict__ g,
                                                 const float* __restrict__ bb,
                                                 const float* __restrict__ add,
                                                 u16* __restrict__ y,
                                                 u16* __restrict__ yp) {
  int row = blockIdx.x;
  int tid = threadIdx.x;
  int lane = tid & 31, wave = tid >> 5;
  const float* xr = x + (size_t)row * D_MODEL;

  float v[4], s = 0.f, s2 = 0.f;
  for (int i = 0; i < 4; ++i) {
    float t = xr[tid + i * 256];
    v[i] = t; s += t; s2 += t * t;
  }
  for (int off = 16; off > 0; off >>= 1) {
    s  += __shfl_xor(s,  off, 32);
    s2 += __shfl_xor(s2, off, 32);
  }
  __shared__ float rs[8], rs2[8];
  if (lane == 0) { rs[wave] = s; rs2[wave] = s2; }
  __syncthreads();
  float S = 0.f, S2 = 0.f;
  for (int w = 0; w < 8; ++w) { S += rs[w]; S2 += rs2[w]; }
  float mu = S * (1.f / D_MODEL);
  float var = S2 * (1.f / D_MODEL) - mu * mu;
  float rstd = rsqrtf(var + 1e-5f);

  for (int i = 0; i < 4; ++i) {
    int idx = tid + i * 256;
    float t = (v[i] - mu) * rstd * g[idx] + bb[idx];
    size_t gi = (size_t)row * D_MODEL + idx;
    y[gi] = f2bf(t);
    if (yp) yp[gi] = f2bf(t + add[gi]);
  }
}

// ---------------- WMMA GEMM: out[M,N] = A[M,K] * B[N,K]^T (+bias)(+relu)(+resid) ----------------
// block = 256 threads (8 waves, 4(M) x 2(N)); block tile 128x128; wave tile 32x64.
// TDM double-buffers the 128x32 A/B k-slices in LDS; WMMA reads fragments from LDS.
__global__ __launch_bounds__(256) void gemm_bf16_nt(const u16* __restrict__ A,
                                                    const u16* __restrict__ Bw,
                                                    int M, int N, int K,
                                                    const float* __restrict__ bias, int relu,
                                                    const float* __restrict__ resid,
                                                    float* __restrict__ outf,
                                                    u16* __restrict__ outb) {
  int tid = threadIdx.x;
  int lane = tid & 31, wave = tid >> 5;
  int wm = wave >> 1, wn = wave & 1;
  int r = lane & 15, c = lane >> 4;
  int mblk = blockIdx.y * 128;
  int nblk = blockIdx.x * 128;
  int m0 = mblk + wm * 32;
  int n0 = nblk + wn * 64;

  v8f acc[2][4];
  for (int i = 0; i < 2; ++i)
    for (int j = 0; j < 4; ++j) acc[i][j] = zero8();

#if HAVE_TDM
  __shared__ u16 As[2][128 * 32];  // 8 KB per buffer
  __shared__ u16 Bs[2][128 * 32];

  if (tid == 0) {
    tdm_load_2d(A  + (size_t)mblk * K, &As[0][0], 128, 32, K);
    tdm_load_2d(Bw + (size_t)nblk * K, &Bs[0][0], 128, 32, K);
    __builtin_amdgcn_s_wait_tensorcnt(0);
  }
  __syncthreads();

  int p = 0;
  for (int k0 = 0; k0 < K; k0 += 32) {
    if (tid == 0 && (k0 + 32) < K) {   // prefetch next k-slice into the other buffer
      tdm_load_2d(A  + (size_t)mblk * K + k0 + 32, &As[1 - p][0], 128, 32, K);
      tdm_load_2d(Bw + (size_t)nblk * K + k0 + 32, &Bs[1 - p][0], 128, 32, K);
    }
    const u16* as = &As[p][0];
    const u16* bs = &Bs[p][0];
    v16bf af[2], bfr[4];
    for (int i = 0; i < 2; ++i)
      af[i] = load_frag_a(as + (wm * 32 + i * 16 + r) * 32 + c * 8);
    for (int j = 0; j < 4; ++j)
      bfr[j] = load_frag_b(bs + (wn * 64 + j * 16 + r) * 32 + c * 8);
    for (int i = 0; i < 2; ++i)
      for (int j = 0; j < 4; ++j)
        acc[i][j] = wmma_bf16(af[i], bfr[j], acc[i][j]);

    if (tid == 0) __builtin_amdgcn_s_wait_tensorcnt(0);
    __syncthreads();
    p ^= 1;
  }
#else
  const u16* a0p = A  + (size_t)(m0 + r) * K + c * 8;
  const u16* a1p = a0p + (size_t)16 * K;
  const u16* bp[4];
  for (int j = 0; j < 4; ++j) bp[j] = Bw + (size_t)(n0 + j * 16 + r) * K + c * 8;

  for (int k0 = 0; k0 < K; k0 += 32) {
    __builtin_prefetch(a0p + k0 + 256, 0, 3);
    __builtin_prefetch(bp[0] + k0 + 256, 0, 3);
    v16bf a0 = load_frag_a(a0p + k0);
    v16bf a1 = load_frag_a(a1p + k0);
    v16bf bfr[4];
    for (int j = 0; j < 4; ++j) bfr[j] = load_frag_b(bp[j] + k0);
    for (int j = 0; j < 4; ++j) {
      acc[0][j] = wmma_bf16(a0, bfr[j], acc[0][j]);
      acc[1][j] = wmma_bf16(a1, bfr[j], acc[1][j]);
    }
  }
#endif

  // D layout: lane (r, half=c) vgpr v -> element (M = c*8+v, N = r)
  for (int i = 0; i < 2; ++i)
    for (int j = 0; j < 4; ++j)
      for (int v = 0; v < 8; ++v) {
        int row = m0 + i * 16 + c * 8 + v;
        int col = n0 + j * 16 + r;
        float x = acc[i][j][v];
        if (bias)  x += bias[col];
        if (relu)  x = fmaxf(x, 0.f);
        if (resid) x += resid[(size_t)row * N + col];
        size_t gi = (size_t)row * N + col;
        if (outf) outf[gi] = x;
        if (outb) outb[gi] = f2bf(x);
      }
}

// ---------------- flash attention (bf16 in, bf16 out), per-wave 16-row l-tile of one (b,h) ----------------
// Q,K,V,O are [T, B, H*64] bf16; row t = l*B + b.
__global__ __launch_bounds__(128) void attn_kernel(const u16* __restrict__ Q,
                                                   const u16* __restrict__ Kx,
                                                   const u16* __restrict__ V,
                                                   u16* __restrict__ O) {
  __shared__ u16 pbuf[4][16 * 32];
  int tid = threadIdx.x;
  int lane = tid & 31, wave = tid >> 5;
  int l0 = (blockIdx.x * 4 + wave) * 16;
  int bh = blockIdx.y;
  int h = bh & (NHEAD - 1), b = bh >> 4;
  int r = lane & 15, c = lane >> 4;
  size_t colbase = (size_t)h * HEADD;
  u16* pb = pbuf[wave];

  // Q A-fragments for K-dim = 64 (2 chunks of 32)
  v16bf qf[2];
  for (int kk = 0; kk < 2; ++kk)
    qf[kk] = load_frag_a(Q + ((size_t)(l0 + r) * BATCH + b) * D_MODEL + colbase + kk * 32 + c * 8);

  float mi[8], li[8], scale_[8];
  v8f oacc[4];
  for (int v = 0; v < 8; ++v) { mi[v] = -1e30f; li[v] = 0.f; }
  for (int t = 0; t < 4; ++t) oacc[t] = zero8();

  for (int s0 = 0; s0 < SEQS; s0 += 32) {
    // scores tile 16 x 32 = two 16x16 D-tiles
    v8f st[2];
    for (int j = 0; j < 2; ++j) {
      v8f sacc = zero8();
      for (int kk = 0; kk < 2; ++kk) {
        v16bf kf = load_frag_b(Kx + ((size_t)(s0 + j * 16 + r) * BATCH + b) * D_MODEL +
                               colbase + kk * 32 + c * 8);
        sacc = wmma_bf16(qf[kk], kf, sacc);
      }
      st[j] = sacc;
    }
    // online softmax over rows (row m = c*8+v spans 16 lanes within half)
    for (int v = 0; v < 8; ++v) {
      st[0][v] *= 0.125f;  // 1/sqrt(64)
      st[1][v] *= 0.125f;
      float t = fmaxf(st[0][v], st[1][v]);
      for (int off = 8; off > 0; off >>= 1) t = fmaxf(t, __shfl_xor(t, off, 16));
      float mn = fmaxf(mi[v], t);
      float sc = __expf(mi[v] - mn);
      float p0 = __expf(st[0][v] - mn);
      float p1 = __expf(st[1][v] - mn);
      st[0][v] = p0; st[1][v] = p1;
      float rsum = p0 + p1;
      for (int off = 8; off > 0; off >>= 1) rsum += __shfl_xor(rsum, off, 16);
      li[v] = li[v] * sc + rsum;
      mi[v] = mn;
      scale_[v] = sc;
    }
    for (int t = 0; t < 4; ++t)
      for (int v = 0; v < 8; ++v) oacc[t][v] *= scale_[v];

    // P (f32 D-layout) -> LDS -> bf16 A-fragment layout
    for (int j = 0; j < 2; ++j)
      for (int v = 0; v < 8; ++v)
        pb[(c * 8 + v) * 32 + j * 16 + r] = f2bf(st[j][v]);
    v16bf pf;
    for (int i = 0; i < 8; ++i) {
      pf[i]     = __builtin_bit_cast(__bf16, pb[r * 32 + c * 8 + i]);
      pf[8 + i] = __builtin_bit_cast(__bf16, pb[r * 32 + 16 + c * 8 + i]);
    }

    // O += P(16x32) * V(32x64): 4 d-tiles; B-frag lane n=r holds V[s0+(i<8 ? c*8+i : 8+c*8+i)][dbase+r]
    for (int t = 0; t < 4; ++t) {
      v16bf vf;
      for (int i = 0; i < 16; ++i)
        vf[i] = __builtin_bit_cast(
            __bf16, V[((size_t)(s0 + (i < 8 ? c * 8 + i : 8 + c * 8 + i)) * BATCH + b) * D_MODEL + colbase + t * 16 + r]);
      oacc[t] = wmma_bf16(pf, vf, oacc[t]);
    }
  }

  for (int t = 0; t < 4; ++t)
    for (int v = 0; v < 8; ++v) {
      float val = oacc[t][v] * (1.f / li[v]);
      O[((size_t)(l0 + c * 8 + v) * BATCH + b) * D_MODEL + colbase + t * 16 + r] = f2bf(val);
    }
}

// ---------------- host-side orchestration ----------------
extern "C" void kernel_launch(void* const* d_in, const int* in_sizes, int n_in,
                              void* d_out, int out_size, void* d_ws, size_t ws_size,
                              hipStream_t stream) {
  (void)in_sizes; (void)n_in; (void)out_size; (void)ws_size;
  const float* tgt       = (const float*)d_in[0];
  const float* memory    = (const float*)d_in[1];
  const float* pos       = (const float*)d_in[2];
  const float* query_pos = (const float*)d_in[3];
  const float* w_sa[4]   = {(const float*)d_in[4], (const float*)d_in[5],
                            (const float*)d_in[6], (const float*)d_in[7]};
  const float* w_ca[4]   = {(const float*)d_in[8], (const float*)d_in[9],
                            (const float*)d_in[10], (const float*)d_in[11]};
  const float* w1 = (const float*)d_in[12];
  const float* b1 = (const float*)d_in[13];
  const float* w2 = (const float*)d_in[14];
  const float* b2 = (const float*)d_in[15];
  const float* ln1_g = (const float*)d_in[16];
  const float* ln1_b = (const float*)d_in[17];
  const float* ln2_g = (const float*)d_in[18];
  const float* ln2_b = (const float*)d_in[19];
  const float* ln3_g = (const float*)d_in[20];
  const float* ln3_b = (const float*)d_in[21];
  const float* alphas = (const float*)d_in[22];

  char* base = (char*)d_ws;
  size_t off = 0;
  auto alloc = [&](size_t bytes) -> void* {
    void* p = base + off;
    off = (off + bytes + 255) & ~(size_t)255;
    return p;
  };

  u16* Wq[10];
  for (int i = 0; i < 8; ++i) Wq[i] = (u16*)alloc((size_t)D_MODEL * D_MODEL * 2);
  Wq[8] = (u16*)alloc((size_t)DFF * D_MODEL * 2);
  Wq[9] = (u16*)alloc((size_t)D_MODEL * DFF * 2);

  u16* t2   = (u16*)alloc((size_t)NTOK * D_MODEL * 2);
  u16* qk   = (u16*)alloc((size_t)NTOK * D_MODEL * 2);
  u16* qb   = (u16*)alloc((size_t)NTOK * D_MODEL * 2);
  u16* kb   = (u16*)alloc((size_t)NTOK * D_MODEL * 2);
  u16* vb   = (u16*)alloc((size_t)NTOK * D_MODEL * 2);
  u16* ctx  = (u16*)alloc((size_t)NTOK * D_MODEL * 2);
  u16* memk = (u16*)alloc((size_t)NMEM * D_MODEL * 2);
  u16* memv = (u16*)alloc((size_t)NMEM * D_MODEL * 2);
  u16* hbuf = (u16*)alloc((size_t)NTOK * DFF * 2);
  float* tgt1  = (float*)alloc((size_t)NTOK * D_MODEL * 4);
  float* tgt2r = (float*)alloc((size_t)NTOK * D_MODEL * 4);

  // 1) fake-quantize all 10 weight matrices to bf16
  int nw = D_MODEL * D_MODEL;
  for (int i = 0; i < 4; ++i)
    quant_kernel<<<nw / 256, 256, 0, stream>>>(w_sa[i], alphas, i, Wq[i], nw);
  for (int i = 0; i < 4; ++i)
    quant_kernel<<<nw / 256, 256, 0, stream>>>(w_ca[i], alphas, 4 + i, Wq[4 + i], nw);
  quant_kernel<<<(DFF * D_MODEL) / 256, 256, 0, stream>>>(w1, alphas, 8, Wq[8], DFF * D_MODEL);
  quant_kernel<<<(D_MODEL * DFF) / 256, 256, 0, stream>>>(w2, alphas, 9, Wq[9], D_MODEL * DFF);

  dim3 gP(D_MODEL / 128, NTOK / 128);   // [4096,1024] GEMMs
  dim3 gF1(DFF / 128, NTOK / 128);      // [4096,4096] GEMM
  dim3 gAttn(SEQL / 16 / 4, BATCH * NHEAD);

  // 2) self-attention block (pre-norm)
  ln_kernel<<<NTOK, 256, 0, stream>>>(tgt, ln1_g, ln1_b, query_pos, t2, qk);
  gemm_bf16_nt<<<gP, 256, 0, stream>>>(qk, Wq[0], NTOK, D_MODEL, D_MODEL, nullptr, 0, nullptr, nullptr, qb);
  gemm_bf16_nt<<<gP, 256, 0, stream>>>(qk, Wq[1], NTOK, D_MODEL, D_MODEL, nullptr, 0, nullptr, nullptr, kb);
  gemm_bf16_nt<<<gP, 256, 0, stream>>>(t2, Wq[2], NTOK, D_MODEL, D_MODEL, nullptr, 0, nullptr, nullptr, vb);
  attn_kernel<<<gAttn, 128, 0, stream>>>(qb, kb, vb, ctx);
  gemm_bf16_nt<<<gP, 256, 0, stream>>>(ctx, Wq[3], NTOK, D_MODEL, D_MODEL, nullptr, 0, tgt, tgt1, nullptr);

  // 3) cross-attention block
  ln_kernel<<<NTOK, 256, 0, stream>>>(tgt1, ln2_g, ln2_b, query_pos, t2, qk);
  addcast_kernel<<<(NMEM * D_MODEL) / 256, 256, 0, stream>>>(memory, pos, memk, NMEM * D_MODEL);
  addcast_kernel<<<(NMEM * D_MODEL) / 256, 256, 0, stream>>>(memory, nullptr, memv, NMEM * D_MODEL);
  gemm_bf16_nt<<<gP, 256, 0, stream>>>(qk, Wq[4], NTOK, D_MODEL, D_MODEL, nullptr, 0, nullptr, nullptr, qb);
  gemm_bf16_nt<<<gP, 256, 0, stream>>>(memk, Wq[5], NMEM, D_MODEL, D_MODEL, nullptr, 0, nullptr, nullptr, kb);
  gemm_bf16_nt<<<gP, 256, 0, stream>>>(memv, Wq[6], NMEM, D_MODEL, D_MODEL, nullptr, 0, nullptr, nullptr, vb);
  attn_kernel<<<gAttn, 128, 0, stream>>>(qb, kb, vb, ctx);
  gemm_bf16_nt<<<gP, 256, 0, stream>>>(ctx, Wq[7], NTOK, D_MODEL, D_MODEL, nullptr, 0, tgt1, tgt2r, nullptr);

  // 4) quantized FFN
  ln_kernel<<<NTOK, 256, 0, stream>>>(tgt2r, ln3_g, ln3_b, nullptr, t2, nullptr);
  gemm_bf16_nt<<<gF1, 256, 0, stream>>>(t2, Wq[8], NTOK, DFF, D_MODEL, b1, 1, nullptr, nullptr, hbuf);
  gemm_bf16_nt<<<gP, 256, 0, stream>>>(hbuf, Wq[9], NTOK, D_MODEL, DFF, b2, 0, tgt2r, (float*)d_out, nullptr);
}
